// DecoupledBottleneckAttention_28767690949362
// MI455X (gfx1250) — compile-verified, matrix operand order checked by reading, not run on hardware
//
#include <hip/hip_runtime.h>
#include <hip/hip_bf16.h>

#define D_MODEL 2048
#define N_HEAD  16
#define S_LEN   2048
#define B_BATCH 2
#define DH      128   // per-head q/k/v dim (64 sem + 64 geo)

typedef __attribute__((ext_vector_type(16))) _Float16 v16h;
typedef __attribute__((ext_vector_type(8)))  float    v8f;
typedef __attribute__((ext_vector_type(4)))  float    f4;
typedef __attribute__((ext_vector_type(2)))  float    f2;
typedef __attribute__((ext_vector_type(4)))  int      i4;

// ---- CDNA5 async global->LDS staging (guarded; falls back to plain copy) ----
#if defined(__has_builtin)
#  if __has_builtin(__builtin_amdgcn_global_load_async_to_lds_b128) && \
      __has_builtin(__builtin_amdgcn_s_wait_asynccnt)
#    define USE_ASYNC_LDS 1
#  endif
#endif
#ifndef USE_ASYNC_LDS
#  define USE_ASYNC_LDS 0
#endif

__device__ __forceinline__ void async_copy16(const _Float16* g, _Float16* l) {
#if USE_ASYNC_LDS
  __builtin_amdgcn_global_load_async_to_lds_b128((i4*)g, (i4*)l, 0, 0);
#else
  *(f4*)l = *(const f4*)g;
#endif
}
__device__ __forceinline__ void async_wait0() {
#if USE_ASYNC_LDS
  __builtin_amdgcn_s_wait_asynccnt(0);
#endif
}

// Load one 16x32 f16 A/B fragment for v_wmma_f32_16x16x32_f16.
// Per ISA layout: lanes 0-15 hold K[0..7] (VGPR0-3) and K[16..23] (VGPR4-7);
// lanes 16-31 hold K[8..15] and K[24..31]. rowp points at this lane's row.
__device__ __forceinline__ v16h load_frag(const _Float16* rowp, int half) {
  union { v16h v; f4 f[2]; } u;
  u.f[0] = *(const f4*)(rowp + half * 8);
  u.f[1] = *(const f4*)(rowp + 16 + half * 8);
  return u.v;
}

// fp32x4 -> f16x4 into LDS (two 32-bit stores; dst 8B aligned)
__device__ __forceinline__ void cvt_store4(_Float16* p, f4 v) {
  union { _Float16 h[4]; unsigned u[2]; } t;
  t.h[0] = (_Float16)v.x; t.h[1] = (_Float16)v.y;
  t.h[2] = (_Float16)v.z; t.h[3] = (_Float16)v.w;
  ((unsigned*)p)[0] = t.u[0];
  ((unsigned*)p)[1] = t.u[1];
}

#define GK   32   // K step
#define GLDS 40   // padded LDS row stride in f16 (80B = 5*16B, b128-aligned rows)

// C[m,n] = sum_k A[m,k]*W[n,k]; A,W fp32; out f16 scattered into
// head-split layout; head_dim = 1<<hd_shift (64 or 128).
// Software-pipelined: next K-step's tile is loaded into registers while the
// current step's WMMAs run, so global latency hides behind compute.
__global__ __launch_bounds__(256) void proj_gemm_kernel(
    const float* __restrict__ A, const float* __restrict__ W,
    _Float16* __restrict__ Out, int K, int hd_shift, int d_base)
{
  __shared__ _Float16 As[128 * GLDS];
  __shared__ _Float16 Bs[128 * GLDS];
  const int tid = threadIdx.x, wave = tid >> 5, lane = tid & 31;
  const int wm = wave >> 2, wn = wave & 3;       // 2x4 wave grid
  const int half = lane >> 4, l16 = lane & 15;
  const long blockM = (long)blockIdx.y * 128;
  const long blockN = (long)blockIdx.x * 128;

  v8f acc[4][2];
  v8f z = {};
#pragma unroll
  for (int i = 0; i < 4; ++i) { acc[i][0] = z; acc[i][1] = z; }

  // prologue: load first tile into registers
  f4 av[4], bv[4];
#pragma unroll
  for (int i = 0; i < 4; ++i) {
    int e = tid + i * 256;              // 0..1023 over 128 rows x 8 float4
    int row = e >> 3, c4 = e & 7;
    av[i] = *(const f4*)(A + (blockM + row) * (long)K + c4 * 4);
    bv[i] = *(const f4*)(W + (blockN + row) * (long)K + c4 * 4);
  }

  for (int k0 = 0; k0 < K; k0 += GK) {
    __syncthreads();                    // previous compute done reading LDS
#pragma unroll
    for (int i = 0; i < 4; ++i) {       // regs -> LDS (fp32 -> f16)
      int e = tid + i * 256;
      int row = e >> 3, c4 = e & 7;
      cvt_store4(&As[row * GLDS + c4 * 4], av[i]);
      cvt_store4(&Bs[row * GLDS + c4 * 4], bv[i]);
    }
    __syncthreads();                    // LDS tile visible
    if (k0 + GK < K) {                  // issue next tile's loads; consumed
#pragma unroll                          // at next iteration's store phase
      for (int i = 0; i < 4; ++i) {
        int e = tid + i * 256;
        int row = e >> 3, c4 = e & 7;
        av[i] = *(const f4*)(A + (blockM + row) * (long)K + (k0 + GK) + c4 * 4);
        bv[i] = *(const f4*)(W + (blockN + row) * (long)K + (k0 + GK) + c4 * 4);
      }
    }
    if (k0 + 2 * GK < K) {              // warm L2 one more step out
      int prow = tid >> 1, pc = (tid & 1) * 16;
      __builtin_prefetch(A + (blockM + prow) * (long)K + k0 + 2 * GK + pc, 0, 3);
      __builtin_prefetch(W + (blockN + prow) * (long)K + k0 + 2 * GK + pc, 0, 3);
    }
    v16h af[4];
#pragma unroll
    for (int i = 0; i < 4; ++i)
      af[i] = load_frag(&As[(wm * 64 + i * 16 + l16) * GLDS], half);
#pragma unroll
    for (int jj = 0; jj < 2; ++jj) {
      v16h bf = load_frag(&Bs[(wn * 32 + jj * 16 + l16) * GLDS], half);
#pragma unroll
      for (int i = 0; i < 4; ++i)
        acc[i][jj] = __builtin_amdgcn_wmma_f32_16x16x32_f16(
            false, af[i], false, bf, (short)0, acc[i][jj], false, false);
    }
  }

  const int hd_mask = (1 << hd_shift) - 1;
#pragma unroll
  for (int i = 0; i < 4; ++i) {
#pragma unroll
    for (int jj = 0; jj < 2; ++jj) {
#pragma unroll
      for (int r = 0; r < 8; ++r) {
        long m = blockM + wm * 64 + i * 16 + half * 8 + r;
        long n = blockN + wn * 32 + jj * 16 + l16;
        int h = (int)(n >> hd_shift);
        int d = d_base + (int)(n & hd_mask);
        long b = m >> 11, s = m & (S_LEN - 1);
        Out[((b * N_HEAD + h) * (long)S_LEN + s) * DH + d] = (_Float16)acc[i][jj][r];
      }
    }
  }
}

// RoPE on the geo half [64..128) of q and k, layout (b,h,s,128)
__global__ void rope_kernel(_Float16* __restrict__ q, _Float16* __restrict__ k)
{
  const int idx = blockIdx.x;              // bh*S + s
  const int s = idx & (S_LEN - 1);
  const int i = threadIdx.x & 31;          // freq index 0..31
  _Float16* base = (threadIdx.x < 32 ? q : k) + (size_t)idx * DH + 64;
  // inv_freq = 10000^(-2i/64) = exp(-i/32 * ln(10000))
  float inv_freq = __expf(-(float)i * (9.210340371976184f / 32.0f));
  float ang = (float)s * inv_freq;
  float sn, cs;
  __sincosf(ang, &sn, &cs);
  float x1 = (float)base[i], x2 = (float)base[i + 32];
  base[i]      = (_Float16)(x1 * cs - x2 * sn);
  base[i + 32] = (_Float16)(x1 * sn + x2 * cs);
}

// Flash attention: block = 4 waves = 64 queries for one (b,h); key tiles of 64.
// Wave owns 16 query rows; online softmax stats live in the WMMA C layout
// (VGPR r <-> rows r / r+8 per lane-half, lanes <-> key columns).
__global__ __launch_bounds__(128) void attn_kernel(
    const _Float16* __restrict__ q, const _Float16* __restrict__ k,
    const _Float16* __restrict__ v, _Float16* __restrict__ o)
{
  __shared__ _Float16 Qs[64 * 136];   // 64 x 128, stride 136 (272B rows)
  __shared__ _Float16 Ks[64 * 136];
  __shared__ _Float16 Vt[128 * 72];   // V transposed: [d][key], stride 72
  __shared__ _Float16 Ps[4 * 16 * 72];// per-wave P tile 16x64, stride 72

  const int tid = threadIdx.x, wave = tid >> 5, lane = tid & 31;
  const int half = lane >> 4, l16 = lane & 15;
  const int qtile = blockIdx.x;
  const int bh = blockIdx.y;
  const size_t bh_off = (size_t)bh * S_LEN * DH;
  const int qbase = qtile * 64;

#pragma unroll
  for (int i = 0; i < 8; ++i) {
    int e = tid + i * 128;              // 64 rows x 16 chunks of 8 f16
    int row = e >> 4, ch = e & 15;
    async_copy16(q + bh_off + (size_t)(qbase + row) * DH + ch * 8,
                 &Qs[row * 136 + ch * 8]);
  }

  float mrow[8], lrow[8];
  v8f oacc[8];
  v8f z = {};
#pragma unroll
  for (int r = 0; r < 8; ++r) { mrow[r] = -1e30f; lrow[r] = 0.f; oacc[r] = z; }

  const float scale = 0.08838834764831845f;   // 1/sqrt(128)

  for (int j = 0; j <= qtile; ++j) {
    __syncthreads();
#pragma unroll
    for (int i = 0; i < 8; ++i) {
      int e = tid + i * 128;
      int row = e >> 4, ch = e & 15;
      async_copy16(k + bh_off + (size_t)(j * 64 + row) * DH + ch * 8,
                   &Ks[row * 136 + ch * 8]);
    }
    // V tile: load all 16 chunks into regs first (pipelined loads), then
    // scatter-transpose into Vt[d][key]
    f2 vv[16];
#pragma unroll
    for (int i = 0; i < 16; ++i) {
      int e = tid + i * 128;            // 64 keys x 32 d-quads
      int key = e >> 5, dq = (e & 31) * 4;
      vv[i] = *(const f2*)(v + bh_off + (size_t)(j * 64 + key) * DH + dq);
    }
#pragma unroll
    for (int i = 0; i < 16; ++i) {
      int e = tid + i * 128;
      int key = e >> 5, dq = (e & 31) * 4;
      union { f2 f; _Float16 h[4]; } u;
      u.f = vv[i];
      Vt[(dq + 0) * 72 + key] = u.h[0];
      Vt[(dq + 1) * 72 + key] = u.h[1];
      Vt[(dq + 2) * 72 + key] = u.h[2];
      Vt[(dq + 3) * 72 + key] = u.h[3];
    }
    async_wait0();                      // asyncs (Q on first iter, K) in LDS
    __syncthreads();

    // S = Q @ K^T : 16 queries x 64 keys, d=128 in 4 K-steps of 32
    v8f sc[4];
#pragma unroll
    for (int t = 0; t < 4; ++t) sc[t] = z;
#pragma unroll
    for (int kk = 0; kk < 4; ++kk) {
      v16h af = load_frag(&Qs[(wave * 16 + l16) * 136 + kk * 32], half);
#pragma unroll
      for (int t = 0; t < 4; ++t) {
        v16h bf = load_frag(&Ks[(t * 16 + l16) * 136 + kk * 32], half);
        sc[t] = __builtin_amdgcn_wmma_f32_16x16x32_f16(
            false, af, false, bf, (short)0, sc[t], false, false);
      }
    }

    // online softmax (row reductions across the 16 lanes of each half)
    const bool diag = (j == qtile);
#pragma unroll
    for (int r = 0; r < 8; ++r) {
      const int qrow = qbase + wave * 16 + half * 8 + r;
      float mx = -1e30f;
#pragma unroll
      for (int t = 0; t < 4; ++t) {
        float val = sc[t][r] * scale;
        if (diag && (j * 64 + t * 16 + l16) > qrow) val = -1e30f;
        sc[t][r] = val;
        mx = fmaxf(mx, val);
      }
#pragma unroll
      for (int d = 1; d < 16; d <<= 1) mx = fmaxf(mx, __shfl_xor(mx, d, 32));
      float nm = fmaxf(mrow[r], mx);
      float rsum = 0.f;
#pragma unroll
      for (int t = 0; t < 4; ++t) {
        float p = __expf(sc[t][r] - nm);
        sc[t][r] = p;
        rsum += p;
      }
#pragma unroll
      for (int d = 1; d < 16; d <<= 1) rsum += __shfl_xor(rsum, d, 32);
      float corr = __expf(mrow[r] - nm);
      lrow[r] = lrow[r] * corr + rsum;
      mrow[r] = nm;
#pragma unroll
      for (int n = 0; n < 8; ++n) oacc[n][r] *= corr;
    }

    // write P (f16) to per-wave LDS tile, then O += P @ V
    _Float16* pw = &Ps[wave * 16 * 72];
#pragma unroll
    for (int t = 0; t < 4; ++t)
#pragma unroll
      for (int r = 0; r < 8; ++r)
        pw[(half * 8 + r) * 72 + t * 16 + l16] = (_Float16)sc[t][r];

#pragma unroll
    for (int kk = 0; kk < 2; ++kk) {    // 64 keys in 2 K-steps of 32
      v16h af = load_frag(&pw[l16 * 72 + kk * 32], half);
#pragma unroll
      for (int n = 0; n < 8; ++n) {     // d = 128 in 8 N-subtiles
        v16h bf = load_frag(&Vt[(n * 16 + l16) * 72 + kk * 32], half);
        oacc[n] = __builtin_amdgcn_wmma_f32_16x16x32_f16(
            false, af, false, bf, (short)0, oacc[n], false, false);
      }
    }
  }

  // normalize and store as (b, s, h*128 + d) f16
  const int b = bh >> 4, hh = bh & 15;
#pragma unroll
  for (int r = 0; r < 8; ++r) {
    float inv = 1.0f / lrow[r];
    int srow = qbase + wave * 16 + half * 8 + r;
#pragma unroll
    for (int n = 0; n < 8; ++n) {
      int d = n * 16 + l16;
      o[((size_t)(b * S_LEN + srow)) * 2048 + hh * 128 + d] =
          (_Float16)(oacc[n][r] * inv);
    }
  }
}

// C[m,n] = sum_k A[m,k]*W[n,k]; A f16 (attn out), W fp32 (Wo), C fp32 (d_out)
__global__ __launch_bounds__(256) void out_gemm_kernel(
    const _Float16* __restrict__ A, const float* __restrict__ W,
    float* __restrict__ C, int Ncols, int K)
{
  __shared__ _Float16 As[128 * GLDS];
  __shared__ _Float16 Bs[128 * GLDS];
  const int tid = threadIdx.x, wave = tid >> 5, lane = tid & 31;
  const int wm = wave >> 2, wn = wave & 3;
  const int half = lane >> 4, l16 = lane & 15;
  const long blockM = (long)blockIdx.y * 128;
  const long blockN = (long)blockIdx.x * 128;

  v8f acc[4][2];
  v8f z = {};
#pragma unroll
  for (int i = 0; i < 4; ++i) { acc[i][0] = z; acc[i][1] = z; }

  // prologue: first W tile into registers (A tile goes via async-to-LDS)
  f4 bv[4];
#pragma unroll
  for (int i = 0; i < 4; ++i) {
    int e = tid + i * 256;
    int row = e >> 3, c4 = e & 7;
    bv[i] = *(const f4*)(W + (blockN + row) * (long)K + c4 * 4);
  }

  for (int k0 = 0; k0 < K; k0 += GK) {
    __syncthreads();
#pragma unroll
    for (int i = 0; i < 2; ++i) {       // A: 128 rows x 4 chunks of 8 f16
      int e = tid + i * 256;
      int row = e >> 2, c8 = e & 3;
      async_copy16(A + (blockM + row) * (long)K + k0 + c8 * 8,
                   &As[row * GLDS + c8 * 8]);
    }
#pragma unroll
    for (int i = 0; i < 4; ++i) {       // W regs -> LDS (fp32 -> f16)
      int e = tid + i * 256;
      int row = e >> 3, c4 = e & 7;
      cvt_store4(&Bs[row * GLDS + c4 * 4], bv[i]);
    }
    if (k0 + GK < K) {                  // next W tile, hidden behind compute
#pragma unroll
      for (int i = 0; i < 4; ++i) {
        int e = tid + i * 256;
        int row = e >> 3, c4 = e & 7;
        bv[i] = *(const f4*)(W + (blockN + row) * (long)K + (k0 + GK) + c4 * 4);
      }
    }
    if (k0 + 2 * GK < K) {
      int prow = tid >> 1, pc = (tid & 1) * 16;
      __builtin_prefetch(W + (blockN + prow) * (long)K + k0 + 2 * GK + pc, 0, 3);
    }
    async_wait0();
    __syncthreads();
    v16h af[4];
#pragma unroll
    for (int i = 0; i < 4; ++i)
      af[i] = load_frag(&As[(wm * 64 + i * 16 + l16) * GLDS], half);
#pragma unroll
    for (int jj = 0; jj < 2; ++jj) {
      v16h bf = load_frag(&Bs[(wn * 32 + jj * 16 + l16) * GLDS], half);
#pragma unroll
      for (int i = 0; i < 4; ++i)
        acc[i][jj] = __builtin_amdgcn_wmma_f32_16x16x32_f16(
            false, af[i], false, bf, (short)0, acc[i][jj], false, false);
    }
  }

#pragma unroll
  for (int i = 0; i < 4; ++i)
#pragma unroll
    for (int jj = 0; jj < 2; ++jj)
#pragma unroll
      for (int r = 0; r < 8; ++r) {
        long m = blockM + wm * 64 + i * 16 + half * 8 + r;
        long n = blockN + wn * 32 + jj * 16 + l16;
        C[m * (long)Ncols + n] = acc[i][jj][r];
      }
}

extern "C" void kernel_launch(void* const* d_in, const int* in_sizes, int n_in,
                              void* d_out, int out_size, void* d_ws, size_t ws_size,
                              hipStream_t stream)
{
  const float* x      = (const float*)d_in[0];
  const float* Wq_sem = (const float*)d_in[1];
  const float* Wk_sem = (const float*)d_in[2];
  const float* Wq_geo = (const float*)d_in[3];
  const float* Wk_geo = (const float*)d_in[4];
  const float* Wv     = (const float*)d_in[5];
  const float* Wo     = (const float*)d_in[6];

  const size_t n_qkv = (size_t)B_BATCH * N_HEAD * S_LEN * DH;   // 8M f16 each
  _Float16* qb = (_Float16*)d_ws;
  _Float16* kb = qb + n_qkv;
  _Float16* vb = kb + n_qkv;
  _Float16* ob = vb + n_qkv;   // attn out, (B,S,2048) f16

  dim3 g1(1024 / 128, (B_BATCH * S_LEN) / 128);   // N=1024 projections
  dim3 g2(2048 / 128, (B_BATCH * S_LEN) / 128);   // N=2048 projections

  proj_gemm_kernel<<<g1, 256, 0, stream>>>(x, Wq_sem, qb, D_MODEL, 6, 0);
  proj_gemm_kernel<<<g1, 256, 0, stream>>>(x, Wq_geo, qb, D_MODEL, 6, 64);
  proj_gemm_kernel<<<g1, 256, 0, stream>>>(x, Wk_sem, kb, D_MODEL, 6, 0);
  proj_gemm_kernel<<<g1, 256, 0, stream>>>(x, Wk_geo, kb, D_MODEL, 6, 64);
  proj_gemm_kernel<<<g2, 256, 0, stream>>>(x, Wv,     vb, D_MODEL, 7, 0);

  rope_kernel<<<B_BATCH * N_HEAD * S_LEN, 64, 0, stream>>>(qb, kb);

  attn_kernel<<<dim3(S_LEN / 64, B_BATCH * N_HEAD), 128, 0, stream>>>(qb, kb, vb, ob);

  out_gemm_kernel<<<dim3(2048 / 128, (B_BATCH * S_LEN) / 128), 256, 0, stream>>>(
      ob, Wo, (float*)d_out, 2048, D_MODEL);
}
